// V2VNetFusion_10153302687872
// MI455X (gfx1250) — compile-verified, matrix operand order checked by reading, not run on hardware
//
#include <hip/hip_runtime.h>
#include <math.h>

#define HH 80
#define WW 80
#define HWP 6400
#define PW 82            // padded width (1-px zero halo each side)
#define PHW 6724         // 82*82
#define CCH 64
#define LL 4
#define NPAIR 32

typedef __attribute__((ext_vector_type(8)))  float        v8f;
typedef __attribute__((ext_vector_type(8)))  unsigned int v8u;
typedef __attribute__((ext_vector_type(16))) __bf16       v16bf;

__device__ __forceinline__ float bf2f(unsigned short u) {
  unsigned int x = ((unsigned int)u) << 16;
  return __builtin_bit_cast(float, x);
}
__device__ __forceinline__ unsigned short f2bf(float f) {
  unsigned int x = __builtin_bit_cast(unsigned int, f);
  unsigned int r = x + 0x7FFFu + ((x >> 16) & 1u);   // round-to-nearest-even
  return (unsigned short)(r >> 16);
}

__device__ __forceinline__ v8f wmma_bf16(v8u a, v8u b, v8f c) {
  return __builtin_amdgcn_wmma_f32_16x16x32_bf16(
      false, __builtin_bit_cast(v16bf, a),
      false, __builtin_bit_cast(v16bf, b),
      (short)0, c, false, false);
}

// A-operand: 16(M=pixels) x 32(K=channels) bf16 from padded NHWC tensor.
// basep = per-lane pixel base (dword ptr); off_dw must be a compile-time
// constant at the call site (fully unrolled) -> immediate-offset b128 loads.
// ISA 16-bit A layout: lane m=lane&15, group g=lane>>4, VGPR r holds
// K = (r>>2)*16 + g*8 + (r&3)*2 (+1)  => dword index (r>>2)*8 + g*4 + (r&3).
__device__ __forceinline__ v8u loadA_off(const unsigned int* basep, int off_dw, int g) {
  v8u a;
#pragma unroll
  for (int r = 0; r < 8; ++r) {
    int ki = ((r >> 2) << 3) + (g << 2) + (r & 3);
    a[r] = basep[off_dw + ki];
  }
  return a;
}

// B-operand: per-lane base + chunk*256 dwords (constant at unrolled call site).
__device__ __forceinline__ v8u loadB_off(const unsigned int* wpl, int chunk) {
  v8u b;
#pragma unroll
  for (int r = 0; r < 8; ++r) b[r] = wpl[chunk * 256 + r];
  return b;
}

// ---------------- zero fill (halo init) ----------------
__global__ void fill_zero(uint4* __restrict__ p, int n) {
  int idx = blockIdx.x * blockDim.x + threadIdx.x;
  if (idx < n) p[idx] = make_uint4(0u, 0u, 0u, 0u);
}

// ---------------- weight packing ----------------
__global__ void pack_weights(const float* __restrict__ src, unsigned int* __restrict__ dst,
                             int O_off, int I_src, int Cin_off, int nCblk, int taps, int total) {
  int idx = blockIdx.x * blockDim.x + threadIdx.x;
  if (idx >= total) return;
  int nchunks = taps * nCblk;
  int r = idx & 7;
  int lane = (idx >> 3) & 31;
  int rest = idx >> 8;
  int chunk = rest % nchunks;
  int ntile = rest / nchunks;
  int n = ntile * 16 + (lane & 15);
  int g = lane >> 4;
  int k0 = ((r >> 2) << 4) + (g << 3) + ((r & 3) << 1);
  int tap = chunk / nCblk;
  int cin = (chunk % nCblk) * 32 + k0;
  float f0 = src[((size_t)(O_off + n) * I_src + (Cin_off + cin)) * taps + tap];
  float f1 = src[((size_t)(O_off + n) * I_src + (Cin_off + cin + 1)) * taps + tap];
  dst[idx] = (unsigned int)f2bf(f0) | ((unsigned int)f2bf(f1) << 16);
}

// ---------------- affine inverse prep ----------------
__global__ void prep_minv(const float* __restrict__ pm, float* __restrict__ minv) {
  int n = threadIdx.x;
  if (n >= NPAIR) return;
  const float* q = pm + n * 16;  // 4x4 row-major
  float a = q[0], b = q[1], tx = q[3] * 1.25f;   // /(0.4*2)
  float d = q[4], e = q[5], ty = q[7] * 1.25f;
  float cx = 40.f, cy = 40.f;
  float t0 = cx - (a * cx + b * cy) + tx;
  float t1 = cy - (d * cx + e * cy) + ty;
  float det = a * e - b * d;
  float ia = e / det, ib = -b / det, ic = -d / det, id = a / det;
  float* o = minv + n * 6;
  o[0] = ia; o[1] = ib; o[2] = -(ia * t0 + ib * t1);
  o[3] = ic; o[4] = id; o[5] = -(ic * t0 + id * t1);
}

// ---------------- nearest mask (dense 80x80 layout) ----------------
__global__ void mask_kernel(const float* __restrict__ minv, float* __restrict__ mask) {
  int idx = blockIdx.x * blockDim.x + threadIdx.x;
  if (idx >= NPAIR * HWP) return;
  int pair = idx / HWP, p = idx - pair * HWP;
  int h = p / WW, w = p - h * WW;
  const float* mv = minv + pair * 6;
  float sx = mv[0] * w + mv[1] * h + mv[2];
  float sy = mv[3] * w + mv[4] * h + mv[5];
  int ix = (int)rintf(sx), iy = (int)rintf(sy);
  mask[idx] = (ix >= 0 && ix < WW && iy >= 0 && iy < HH) ? 1.0f : 0.0f;
}

// ------- rotate NCHW f32 -> rotated padded NHWC bf16: y[c,h,w]=x[c,W-1-w,h] --
__global__ void rotate_in(const float* __restrict__ x, unsigned short* __restrict__ cat) {
  int idx = blockIdx.x * blockDim.x + threadIdx.x;  // exact 8*6400*64
  int c = idx & 63;
  int rest = idx >> 6;
  int w = rest % WW;
  int rest2 = rest / WW;
  int h = rest2 % HH;
  int img = rest2 / HH;
  float v = x[(size_t)img * CCH * HWP + (size_t)c * HWP + (WW - 1 - w) * WW + h];
  cat[((size_t)img * PHW + (h + 1) * PW + (w + 1)) * 128 + c] = f2bf(v);
}

// ---------------- bilinear warp into padded nbr ----------------
__global__ void warp_bilinear(const unsigned short* __restrict__ cat,
                              const float* __restrict__ minv,
                              unsigned short* __restrict__ nbr) {
  int idx = blockIdx.x * blockDim.x + threadIdx.x;
  if (idx >= NPAIR * HWP) return;
  int pair = idx / HWP, p = idx - pair * HWP;
  int h = p / WW, w = p - h * WW;
  int b = pair >> 4, i = (pair >> 2) & 3;
  const float* mv = minv + pair * 6;
  float sx = mv[0] * w + mv[1] * h + mv[2];
  float sy = mv[3] * w + mv[4] * h + mv[5];
  float x0f = floorf(sx), y0f = floorf(sy);
  float fx = sx - x0f, fy = sy - y0f;
  int x0 = (int)x0f, y0 = (int)y0f;
  float w00 = (1.f - fx) * (1.f - fy), w10 = fx * (1.f - fy);
  float w01 = (1.f - fx) * fy, w11 = fx * fy;
  bool vx0 = (x0 >= 0) && (x0 < WW), vx1 = (x0 + 1 >= 0) && (x0 + 1 < WW);
  bool vy0 = (y0 >= 0) && (y0 < HH), vy1 = (y0 + 1 >= 0) && (y0 + 1 < HH);
  const unsigned short* src = cat + (size_t)(b * LL + i) * PHW * 128;
  const unsigned short* p00 = src + ((size_t)(y0 + 1) * PW + (x0 + 1)) * 128;
  const unsigned short* p10 = p00 + 128;
  const unsigned short* p01 = p00 + (size_t)PW * 128;
  const unsigned short* p11 = p01 + 128;
  unsigned short* dst = nbr + (size_t)pair * PHW * 64 + ((size_t)(h + 1) * PW + (w + 1)) * 64;
#pragma unroll 4
  for (int c = 0; c < CCH; ++c) {
    float acc = 0.f;
    if (vx0 && vy0) acc += w00 * bf2f(p00[c]);
    if (vx1 && vy0) acc += w10 * bf2f(p10[c]);
    if (vx0 && vy1) acc += w01 * bf2f(p01[c]);
    if (vx1 && vy1) acc += w11 * bf2f(p11[c]);
    dst[c] = f2bf(acc);
  }
}

// ---- fully-unrolled 3x3 conv GEMM core: 16 px x 16 out-ch, K = nCblk*32*9 ---
// basep: per-lane pixel base (dword ptr) at padded (h, w0+m), channel 0.
// wpl:   per-lane packed-B base for this wave's out-channel tile.
template <int Ct, int NCBLK>
__device__ __forceinline__ v8f conv3x3_acc(const unsigned int* basep,
                                           const unsigned int* wpl, int g, v8f acc) {
#pragma unroll
  for (int dy = 0; dy < 3; ++dy)
#pragma unroll
    for (int dx = 0; dx < 3; ++dx)
#pragma unroll
      for (int cblk = 0; cblk < NCBLK; ++cblk) {
        int off = (((dy * PW + dx) * Ct + cblk * 32) >> 1);     // constant
        int chunk = (dy * 3 + dx) * NCBLK + cblk;               // constant
        acc = wmma_bf16(loadA_off(basep, off, g), loadB_off(wpl, chunk), acc);
      }
  return acc;
}

// ---------------- msg conv + masked mean -> agg (cat channels 64..127) ------
__global__ __launch_bounds__(128) void msg_conv(const unsigned short* __restrict__ cat,
                                                const unsigned short* __restrict__ nbr,
                                                const unsigned int* __restrict__ wpn,
                                                const unsigned int* __restrict__ wpe,
                                                const float* __restrict__ mask,
                                                const float* __restrict__ msg_b,
                                                unsigned short* __restrict__ cat_out) {
  int blk = blockIdx.x;
  int imgj = blk / 400, tile = blk - imgj * 400;
  int b = imgj >> 2, j = imgj & 3;
  int h = tile / 5, w0 = (tile % 5) * 16;
  int lane = threadIdx.x & 31, wv = threadIdx.x >> 5;
  int m = lane & 15, g = lane >> 4;

  // ego conv (shared across all 4 neighbors), Ct=128, 2 channel blocks
  const unsigned int* egoP =
      (const unsigned int*)(cat + ((size_t)imgj * PHW + h * PW + w0 + m) * 128);
  const unsigned int* wpeL = wpe + ((size_t)(wv * 18) * 32 + lane) * 8;
  v8f accE = {0.f, 0.f, 0.f, 0.f, 0.f, 0.f, 0.f, 0.f};
  accE = conv3x3_acc<128, 2>(egoP, wpeL, g, accE);

  int n = wv * 16 + m;
  float bias = msg_b[n];
  float agg[8] = {0.f, 0.f, 0.f, 0.f, 0.f, 0.f, 0.f, 0.f};
  const unsigned int* wpnL = wpn + ((size_t)(wv * 18) * 32 + lane) * 8;

  for (int i = 0; i < LL; ++i) {
    int pair = (b * LL + i) * LL + j;
    const unsigned int* nbP =
        (const unsigned int*)(nbr + ((size_t)pair * PHW + h * PW + w0 + m) * 64);
    v8f accN = {0.f, 0.f, 0.f, 0.f, 0.f, 0.f, 0.f, 0.f};
    accN = conv3x3_acc<64, 2>(nbP, wpnL, g, accN);
    const float* mrow = mask + (size_t)pair * HWP + h * WW + w0;
#pragma unroll
    for (int r = 0; r < 8; ++r) {
      float mk = mrow[g * 8 + r];
      agg[r] += mk * (accN[r] + accE[r] + bias);
    }
  }

  unsigned short* outp = cat_out + (size_t)imgj * PHW * 128;
#pragma unroll
  for (int r = 0; r < 8; ++r) {
    int mm = g * 8 + r;
    outp[((size_t)(h + 1) * PW + (w0 + mm + 1)) * 128 + 64 + n] = f2bf(agg[r] * 0.25f);
  }
}

// --------- GRU: update = sigmoid(gatesU), h1 = update * tanh(cand) ----------
// writes h1 directly as next iteration's rotated y (cat_next channels 0..63)
__global__ __launch_bounds__(128) void gru_conv(const unsigned short* __restrict__ cat,
                                                const unsigned int* __restrict__ wpg,
                                                const unsigned int* __restrict__ wpc,
                                                const float* __restrict__ gates_b,
                                                const float* __restrict__ can_b,
                                                unsigned short* __restrict__ cat_next) {
  int blk = blockIdx.x;
  int img = blk / 400, tile = blk - img * 400;
  int h = tile / 5, w0 = (tile % 5) * 16;
  int lane = threadIdx.x & 31, wv = threadIdx.x >> 5;
  int m = lane & 15, g = lane >> 4;
  const unsigned int* srcP =
      (const unsigned int*)(cat + ((size_t)img * PHW + h * PW + w0 + m) * 128);
  const unsigned int* wpgL = wpg + ((size_t)(wv * 36) * 32 + lane) * 8;
  const unsigned int* wpcL = wpc + ((size_t)(wv * 36) * 32 + lane) * 8;
  v8f aG = {0.f, 0.f, 0.f, 0.f, 0.f, 0.f, 0.f, 0.f};
  v8f aC = {0.f, 0.f, 0.f, 0.f, 0.f, 0.f, 0.f, 0.f};
#pragma unroll
  for (int dy = 0; dy < 3; ++dy)
#pragma unroll
    for (int dx = 0; dx < 3; ++dx)
#pragma unroll
      for (int cblk = 0; cblk < 4; ++cblk) {
        int off = (((dy * PW + dx) * 128 + cblk * 32) >> 1);
        int chunk = (dy * 3 + dx) * 4 + cblk;
        v8u a = loadA_off(srcP, off, g);
        aG = wmma_bf16(a, loadB_off(wpgL, chunk), aG);
        aC = wmma_bf16(a, loadB_off(wpcL, chunk), aC);
      }
  int n = wv * 16 + m;
  float bg = gates_b[64 + n], bc = can_b[n];
  unsigned short* outp = cat_next + (size_t)img * PHW * 128;
#pragma unroll
  for (int r = 0; r < 8; ++r) {
    int mm = g * 8 + r;
    float u = 1.0f / (1.0f + expf(-(aG[r] + bg)));
    float hv = u * tanhf(aC[r] + bc);
    outp[((size_t)(h + 1) * PW + (w0 + mm + 1)) * 128 + n] = f2bf(hv);
  }
}

// ---------------- final MLP with un-rotation folded into gather -------------
// out[b,h,w,o] = sum_c catF[b*L][w][79-h][c] * mlp_w[o,c] + mlp_b[o]
__global__ __launch_bounds__(128) void mlp_gemm(const unsigned short* __restrict__ catF,
                                                const unsigned int* __restrict__ wpm,
                                                const float* __restrict__ mlp_b,
                                                float* __restrict__ out) {
  int blk = blockIdx.x;
  int b = blk / 400, tile = blk - b * 400;
  int h = tile / 5, w0 = (tile % 5) * 16;
  int lane = threadIdx.x & 31, wv = threadIdx.x >> 5;
  int m = lane & 15, g = lane >> 4;
  const unsigned int* p =
      (const unsigned int*)(catF + ((size_t)(b * LL) * PHW + (size_t)(w0 + m + 1) * PW + (HH - h)) * 128);
  const unsigned int* wpmL = wpm + ((size_t)(wv * 2) * 32 + lane) * 8;
  v8f acc = {0.f, 0.f, 0.f, 0.f, 0.f, 0.f, 0.f, 0.f};
#pragma unroll
  for (int chunk = 0; chunk < 2; ++chunk)
    acc = wmma_bf16(loadA_off(p, chunk * 16, g), loadB_off(wpmL, chunk), acc);
  int n = wv * 16 + m;
  float bias = mlp_b[n];
#pragma unroll
  for (int r = 0; r < 8; ++r) {
    int mm = g * 8 + r;
    out[((size_t)((b * HH + h) * WW) + w0 + mm) * 64 + n] = acc[r] + bias;
  }
}

extern "C" void kernel_launch(void* const* d_in, const int* in_sizes, int n_in,
                              void* d_out, int out_size, void* d_ws, size_t ws_size,
                              hipStream_t stream) {
  const float* x       = (const float*)d_in[0];
  const float* pm      = (const float*)d_in[2];
  const float* msg_w   = (const float*)d_in[4];
  const float* msg_b   = (const float*)d_in[5];
  const float* gates_w = (const float*)d_in[6];
  const float* gates_b = (const float*)d_in[7];
  const float* can_w   = (const float*)d_in[8];
  const float* can_b   = (const float*)d_in[9];
  const float* mlp_w   = (const float*)d_in[10];
  const float* mlp_b   = (const float*)d_in[11];
  float* out = (float*)d_out;
  (void)in_sizes; (void)n_in; (void)out_size; (void)ws_size;

  char* ws = (char*)d_ws;
  float* minv        = (float*)(ws + 0);              //     768 B
  float* mask        = (float*)(ws + 1024);           //  819200 B
  unsigned int* wpn  = (unsigned int*)(ws + 820224);  //   73728 B
  unsigned int* wpe  = (unsigned int*)(ws + 893952);  //   73728 B
  unsigned int* wpg  = (unsigned int*)(ws + 967680);  //  147456 B
  unsigned int* wpc  = (unsigned int*)(ws + 1115136); //  147456 B
  unsigned int* wpm  = (unsigned int*)(ws + 1262592); //    8192 B
  // padded activation buffers (contiguous so one fill covers all):
  unsigned short* catA = (unsigned short*)(ws + 1270784);                 // 13770752 B
  unsigned short* catB = (unsigned short*)(ws + 1270784 + 13770752);      // 13770752 B
  unsigned short* nbr  = (unsigned short*)(ws + 1270784 + 2 * 13770752);  // 27541504 B

  // zero halos (and everything else) : 55,083,008 B = 3,442,688 uint4
  fill_zero<<<13448, 256, 0, stream>>>((uint4*)catA, 3442688);

  prep_minv<<<1, 32, 0, stream>>>(pm, minv);
  mask_kernel<<<800, 256, 0, stream>>>(minv, mask);
  // msg_w (64,128,3,3): neighbor half cin 0..63, ego half cin 64..127
  pack_weights<<<72, 256, 0, stream>>>(msg_w, wpn, 0, 128, 0, 2, 9, 18432);
  pack_weights<<<72, 256, 0, stream>>>(msg_w, wpe, 0, 128, 64, 2, 9, 18432);
  // gates_w (128,192,3,3): update rows 64..127, cin 0..127 (h0 block is zero)
  pack_weights<<<144, 256, 0, stream>>>(gates_w, wpg, 64, 192, 0, 4, 9, 36864);
  // can_w (64,192,3,3): cin 0..127
  pack_weights<<<144, 256, 0, stream>>>(can_w, wpc, 0, 192, 0, 4, 9, 36864);
  // mlp_w (64,64)
  pack_weights<<<8, 256, 0, stream>>>(mlp_w, wpm, 0, 64, 0, 2, 1, 2048);

  rotate_in<<<12800, 256, 0, stream>>>(x, catA);

  // iteration 0 (rotated frame throughout)
  warp_bilinear<<<800, 256, 0, stream>>>(catA, minv, nbr);
  msg_conv<<<3200, 128, 0, stream>>>(catA, nbr, wpn, wpe, mask, msg_b, catA);
  gru_conv<<<3200, 128, 0, stream>>>(catA, wpg, wpc, gates_b, can_b, catB);

  // iteration 1
  warp_bilinear<<<800, 256, 0, stream>>>(catB, minv, nbr);
  msg_conv<<<3200, 128, 0, stream>>>(catB, nbr, wpn, wpe, mask, msg_b, catB);
  gru_conv<<<3200, 128, 0, stream>>>(catB, wpg, wpc, gates_b, can_b, catA);

  mlp_gemm<<<800, 128, 0, stream>>>(catA, wpm, mlp_b, out);
}
